// SelfAttention_66838281061336
// MI455X (gfx1250) — matrix operand order repaired and checked by reading, not compiled, not run on hardware
//
#include <hip/hip_runtime.h>

typedef _Float16 f16;
typedef __attribute__((ext_vector_type(16))) _Float16 v16h;
typedef __attribute__((ext_vector_type(8)))  _Float16 v8h;
typedef __attribute__((ext_vector_type(8)))  float    v8f;

#define BB     8
#define CCH    256
#define NNP    1024
#define HEADS  8
#define HD     32
#define GROUPS 32
#define CPG    8          // channels per group
#define EPSV   1e-5f
#define QSCALE 0.17677669529663687f   // 1/sqrt(32)

union V16 { v16h v; v8h h[2]; };

// ---------------------------------------------------------------------------
// Kernel 0: convert weights to f16; fold 1/sqrt(HD) into W_q rows.
// ---------------------------------------------------------------------------
__global__ void wconv_kernel(const float* __restrict__ qkv_w,
                             const float* __restrict__ proj_w,
                             f16* __restrict__ qw16, f16* __restrict__ pw16) {
    int i = blockIdx.x * 256 + threadIdx.x;           // 0 .. 262143
    if (i < 3 * CCH * CCH) {
        float v = qkv_w[i];
        if (i < CCH * CCH) v *= QSCALE;               // q rows
        qw16[i] = (f16)v;
    } else {
        int j = i - 3 * CCH * CCH;                    // 0 .. 65535
        pw16[j] = (f16)proj_w[j];
    }
}

// ---------------------------------------------------------------------------
// Kernel 1: GroupNorm. One block per (b, group). Writes h transposed to
// [b][n][c] f16 so QKV B-fragments become contiguous 16B loads.
// ---------------------------------------------------------------------------
__global__ void gn_kernel(const float* __restrict__ x,
                          const float* __restrict__ gw,
                          const float* __restrict__ gb,
                          f16* __restrict__ h16) {
    __shared__ float ssum[256], ssq[256];
    int bid = blockIdx.x;                 // b*GROUPS + g
    int b = bid >> 5, g = bid & 31;
    int t = threadIdx.x;
    const float* xg = x + ((size_t)b * CCH + g * CPG) * NNP;   // 8 x 1024

    float s = 0.f, q = 0.f;
    for (int e = t; e < CPG * NNP; e += 256) {
        float v = xg[e];
        s += v; q += v * v;
    }
    ssum[t] = s; ssq[t] = q;
    __syncthreads();
    for (int off = 128; off > 0; off >>= 1) {
        if (t < off) { ssum[t] += ssum[t + off]; ssq[t] += ssq[t + off]; }
        __syncthreads();
    }
    const float inv_n = 1.f / (float)(CPG * NNP);
    float mu   = ssum[0] * inv_n;
    float var  = ssq[0] * inv_n - mu * mu;
    float rinv = rsqrtf(var + EPSV);

    for (int e = t; e < CPG * NNP; e += 256) {
        int cl = e >> 10, n = e & 1023;
        int c  = g * CPG + cl;
        float v = (xg[e] - mu) * rinv * gw[c] + gb[c];
        h16[((size_t)b * NNP + n) * CCH + c] = (f16)v;
    }
}

// ---------------------------------------------------------------------------
// Kernel 2: QKV GEMM. out[o][n] = sum_c W[o][c] * h[c][n], per batch.
// One wave = one 16(o) x 16(n) tile, K-loop 8 x (K=32) WMMA.
// q,k stored [b][h][n][d]; v stored [b][h][d][n].
// ---------------------------------------------------------------------------
__global__ void qkv_kernel(const f16* __restrict__ w16,
                           const f16* __restrict__ h16,
                           const float* __restrict__ bias,
                           f16* __restrict__ q16, f16* __restrict__ k16,
                           f16* __restrict__ v16) {
    int wid  = blockIdx.x * 8 + (threadIdx.x >> 5);
    int lane = threadIdx.x & 31;
    int b  = wid / (48 * 64);
    int r0 = wid % (48 * 64);
    int mt = r0 >> 6;                      // 0..47  (o tile)
    int nt = r0 & 63;                      // 0..63  (n tile)
    int lhalf = lane & 15;
    int hi16  = lane >> 4;

    // A fragment source: W row-major [o][c]
    const f16* aptr = w16 + (size_t)(mt * 16 + lhalf) * CCH + (hi16 ? 16 : 0);
    // B fragment source: h [n][c] (c contiguous)
    const f16* bptr = h16 + ((size_t)b * NNP + nt * 16 + lhalf) * CCH + (hi16 ? 16 : 0);

    v8f acc = {};
    for (int kt = 0; kt < 8; ++kt) {
        int c0 = kt * 32;
        __builtin_prefetch(aptr + c0 + 64);
        __builtin_prefetch(bptr + c0 + 64);
        V16 A, Bv;
        A.h[0]  = *(const v8h*)(aptr + c0);
        A.h[1]  = *(const v8h*)(aptr + c0 + 8);
        Bv.h[0] = *(const v8h*)(bptr + c0);
        Bv.h[1] = *(const v8h*)(bptr + c0 + 8);
        acc = __builtin_amdgcn_wmma_f32_16x16x32_f16(false, A.v, false, Bv.v,
                                                     (short)0, acc, false, false);
    }

    int mbase = mt * 16 + (hi16 ? 8 : 0);  // first of this lane's 8 output rows
    int sel   = mbase >> 8;                // 0=q 1=k 2=v
    int om    = mbase & 255;
    int head  = om >> 5;
    int dbase = om & 31;                   // multiple of 8
    int n     = nt * 16 + lhalf;

    if (sel == 2) {
        // v layout [b][head][d][n] -> strided scalar stores
        f16* vp = v16 + (((size_t)(b * HEADS + head) * HD + dbase) * NNP) + n;
        #pragma unroll
        for (int r = 0; r < 8; ++r)
            vp[(size_t)r * NNP] = (f16)(acc[r] + bias[mbase + r]);
    } else {
        v8h pk;
        #pragma unroll
        for (int r = 0; r < 8; ++r) {
            float bv = bias[mbase + r];
            if (sel == 0) bv *= QSCALE;    // scale folded into q
            pk[r] = (f16)(acc[r] + bv);
        }
        f16* dst = (sel ? k16 : q16) +
                   (((size_t)(b * HEADS + head) * NNP + n) * HD) + dbase;
        *(v8h*)dst = pk;                   // one 16B packed store
    }
}

// ---------------------------------------------------------------------------
// Kernel 3: fused flash attention. One wave = one (b, head, 16-row i-tile).
// Loops j in chunks of 32: S = Q*K (2 WMMA), online softmax with shfl_xor
// row reductions, P through per-wave LDS (D-layout -> A-layout), O += P*V^T
// (2 WMMA). Output written [b][n][c] f16.
// ---------------------------------------------------------------------------
__global__ void attn_kernel(const f16* __restrict__ q16,
                            const f16* __restrict__ k16,
                            const f16* __restrict__ v16,
                            f16* __restrict__ o16) {
    __shared__ f16 plds[8][16 * 32];       // 8 KB: per-wave 16x32 P tile
    int wave = threadIdx.x >> 5;
    int lane = threadIdx.x & 31;
    int wid  = blockIdx.x * 8 + wave;
    int b    = wid >> 9;                   // / 512
    int r0   = wid & 511;
    int head = r0 >> 6;
    int it   = r0 & 63;
    int lhalf = lane & 15;
    int hi16  = lane >> 4;

    const f16* qbase = q16 + ((size_t)(b * HEADS + head) * NNP) * HD;
    const f16* kbase = k16 + ((size_t)(b * HEADS + head) * NNP) * HD;
    const f16* vbase = v16 + ((size_t)(b * HEADS + head) * HD) * NNP;

    // Q A-fragment (16 x 32), loaded once
    V16 Qf;
    {
        const f16* qr = qbase + (size_t)(it * 16 + lhalf) * HD + (hi16 ? 16 : 0);
        Qf.h[0] = *(const v8h*)qr;
        Qf.h[1] = *(const v8h*)(qr + 8);
    }

    float m[8], l[8];
    v8f o0 = {}, o1 = {};
    #pragma unroll
    for (int r = 0; r < 8; ++r) { m[r] = -3.0e38f; l[r] = 0.f; }

    f16* pl  = &plds[wave][0];
    int prow = hi16 ? 8 : 0;

    for (int jt = 0; jt < NNP / 32; ++jt) {
        // K B-fragments: k [n][d] -> lane = column j, 16 consecutive d
        V16 K0, K1;
        {
            const f16* kr = kbase + (size_t)(jt * 32 + lhalf) * HD + (hi16 ? 16 : 0);
            K0.h[0] = *(const v8h*)kr;
            K0.h[1] = *(const v8h*)(kr + 8);
            kr += (size_t)16 * HD;
            K1.h[0] = *(const v8h*)kr;
            K1.h[1] = *(const v8h*)(kr + 8);
        }
        v8f s0 = {}, s1 = {};
        s0 = __builtin_amdgcn_wmma_f32_16x16x32_f16(false, Qf.v, false, K0.v,
                                                    (short)0, s0, false, false);
        s1 = __builtin_amdgcn_wmma_f32_16x16x32_f16(false, Qf.v, false, K1.v,
                                                    (short)0, s1, false, false);

        // online softmax; D layout: lane holds col N=lhalf, VGPR r -> row prow+r
        #pragma unroll
        for (int r = 0; r < 8; ++r) {
            float mx = fmaxf(s0[r], s1[r]);
            mx = fmaxf(mx, __shfl_xor(mx, 1, 32));
            mx = fmaxf(mx, __shfl_xor(mx, 2, 32));
            mx = fmaxf(mx, __shfl_xor(mx, 4, 32));
            mx = fmaxf(mx, __shfl_xor(mx, 8, 32));
            float mn = fmaxf(m[r], mx);
            float a  = __expf(m[r] - mn);
            float p0 = __expf(s0[r] - mn);
            float p1 = __expf(s1[r] - mn);
            float rs = p0 + p1;
            rs += __shfl_xor(rs, 1, 32);
            rs += __shfl_xor(rs, 2, 32);
            rs += __shfl_xor(rs, 4, 32);
            rs += __shfl_xor(rs, 8, 32);
            l[r] = l[r] * a + rs;
            m[r] = mn;
            o0[r] *= a;
            o1[r] *= a;
            pl[(prow + r) * 32 + lhalf]      = (f16)p0;
            pl[(prow + r) * 32 + 16 + lhalf] = (f16)p1;
        }
        asm volatile("s_wait_dscnt 0x0" ::: "memory");   // P visible wave-wide

        // P A-fragment from LDS
        V16 Pf;
        {
            const f16* pr = pl + lhalf * 32 + (hi16 ? 16 : 0);
            Pf.h[0] = *(const v8h*)pr;
            Pf.h[1] = *(const v8h*)(pr + 8);
        }
        // V B-fragments: v [d][n] -> lane = column d, 16 consecutive j
        V16 V0, V1;
        {
            const f16* vr = vbase + (size_t)lhalf * NNP + jt * 32 + (hi16 ? 16 : 0);
            V0.h[0] = *(const v8h*)vr;
            V0.h[1] = *(const v8h*)(vr + 8);
            vr += (size_t)16 * NNP;
            V1.h[0] = *(const v8h*)vr;
            V1.h[1] = *(const v8h*)(vr + 8);
        }
        o0 = __builtin_amdgcn_wmma_f32_16x16x32_f16(false, Pf.v, false, V0.v,
                                                    (short)0, o0, false, false);
        o1 = __builtin_amdgcn_wmma_f32_16x16x32_f16(false, Pf.v, false, V1.v,
                                                    (short)0, o1, false, false);
        asm volatile("s_wait_dscnt 0x0" ::: "memory");   // LDS WAR safety
    }

    // normalize and store O as [b][i][c], c = head*32 + dtile*16 + lhalf
    #pragma unroll
    for (int r = 0; r < 8; ++r) {
        float inv = 1.f / l[r];
        int i = it * 16 + prow + r;
        size_t base = ((size_t)b * NNP + i) * CCH + head * HD;
        o16[base + lhalf]      = (f16)(o0[r] * inv);
        o16[base + 16 + lhalf] = (f16)(o1[r] * inv);
    }
}

// ---------------------------------------------------------------------------
// Kernel 4: projection GEMM + bias + residual (f32 output).
// ---------------------------------------------------------------------------
__global__ void proj_kernel(const f16* __restrict__ pw16,
                            const f16* __restrict__ o16,
                            const float* __restrict__ pb,
                            const float* __restrict__ x,
                            float* __restrict__ out) {
    int wid  = blockIdx.x * 8 + (threadIdx.x >> 5);
    int lane = threadIdx.x & 31;
    int b  = wid >> 10;                    // / (16*64)
    int r0 = wid & 1023;
    int mt = r0 >> 6;
    int nt = r0 & 63;
    int lhalf = lane & 15;
    int hi16  = lane >> 4;

    const f16* aptr = pw16 + (size_t)(mt * 16 + lhalf) * CCH + (hi16 ? 16 : 0);
    const f16* bptr = o16  + ((size_t)b * NNP + nt * 16 + lhalf) * CCH + (hi16 ? 16 : 0);

    v8f acc = {};
    for (int kt = 0; kt < 8; ++kt) {
        int c0 = kt * 32;
        __builtin_prefetch(aptr + c0 + 64);
        __builtin_prefetch(bptr + c0 + 64);
        V16 A, Bv;
        A.h[0]  = *(const v8h*)(aptr + c0);
        A.h[1]  = *(const v8h*)(aptr + c0 + 8);
        Bv.h[0] = *(const v8h*)(bptr + c0);
        Bv.h[1] = *(const v8h*)(bptr + c0 + 8);
        acc = __builtin_amdgcn_wmma_f32_16x16x32_f16(false, A.v, false, Bv.v,
                                                     (short)0, acc, false, false);
    }

    int mbase = mt * 16 + (hi16 ? 8 : 0);
    int n     = nt * 16 + lhalf;
    #pragma unroll
    for (int r = 0; r < 8; ++r) {
        int o = mbase + r;
        size_t idx = ((size_t)(b * CCH + o)) * NNP + n;
        out[idx] = x[idx] + pb[o] + acc[r];
    }
}

// ---------------------------------------------------------------------------
extern "C" void kernel_launch(void* const* d_in, const int* in_sizes, int n_in,
                              void* d_out, int out_size, void* d_ws, size_t ws_size,
                              hipStream_t stream) {
    const float* x      = (const float*)d_in[0];
    const float* gn_w   = (const float*)d_in[1];
    const float* gn_b   = (const float*)d_in[2];
    const float* qkv_w  = (const float*)d_in[3];
    const float* qkv_b  = (const float*)d_in[4];
    const float* proj_w = (const float*)d_in[5];
    const float* proj_b = (const float*)d_in[6];
    float* out = (float*)d_out;

    char* w = (char*)d_ws;
    f16* h16  = (f16*)(w);                          // 4 MB, reused as o16 later
    f16* o16  = h16;
    f16* q16  = (f16*)(w + (4  << 20));             // 4 MB
    f16* k16  = (f16*)(w + (8  << 20));             // 4 MB
    f16* v16  = (f16*)(w + (12 << 20));             // 4 MB
    f16* qw16 = (f16*)(w + (16 << 20));             // 384 KB
    f16* pw16 = (f16*)(w + (16 << 20) + 3 * CCH * CCH * 2);  // 128 KB

    // 0: weight convert (262144 elements)
    wconv_kernel<<<1024, 256, 0, stream>>>(qkv_w, proj_w, qw16, pw16);
    // 1: group norm (B*GROUPS blocks)
    gn_kernel<<<BB * GROUPS, 256, 0, stream>>>(x, gn_w, gn_b, h16);
    // 2: QKV GEMM: 8*48*64 tiles / 8 waves
    qkv_kernel<<<3072, 256, 0, stream>>>(qw16, h16, qkv_b, q16, k16, v16);
    // 3: flash attention: 8*8*64 waves / 8
    attn_kernel<<<512, 256, 0, stream>>>(q16, k16, v16, o16);
    // 4: proj + residual: 8*16*64 tiles / 8
    proj_kernel<<<1024, 256, 0, stream>>>(pw16, o16, proj_b, x, out);
}
